// MA_LSTM_45191645888982
// MI455X (gfx1250) — compile-verified, matrix operand order checked by reading, not run on hardware
//
#include <hip/hip_runtime.h>
#include <math.h>

#define DEV __device__ __forceinline__

typedef __attribute__((ext_vector_type(16))) __bf16 v16bf;
typedef __attribute__((ext_vector_type(8)))  __bf16 v8bf;
typedef __attribute__((ext_vector_type(8)))  float  v8f;

constexpr int U  = 1024;
constexpr int B  = 64;
constexpr int T  = 128;
constexpr int D  = 256;
constexpr int U5 = 5 * U;   // 5120
constexpr int U3 = 3 * U;   // 3072

// ---------------------------------------------------------------------------
// WMMA helpers (CDNA5 16-bit A/B fragment layout, wave32)
//   lane L: row/col index = L & 15, half = L >> 4
//   frag elems 0..7  hold K = k0 + half*8 + i
//   frag elems 8..15 hold K = k0 + 16 + half*8 + i
// ---------------------------------------------------------------------------
DEV v16bf ldfrag(const __bf16* row, int k0, int half) {
    const __bf16* p = row + k0 + half * 8;
    v8bf lo = *(const v8bf*)(p);
    v8bf hi = *(const v8bf*)(p + 16);
    return __builtin_shufflevector(lo, hi, 0,1,2,3,4,5,6,7,8,9,10,11,12,13,14,15);
}

DEV v8f wmma_bf16(v16bf a, v16bf b, v8f c) {
    // (neg_a, A, neg_b, B, c_mod, C, reuse_a, reuse_b)
    return __builtin_amdgcn_wmma_f32_16x16x32_bf16(false, a, false, b, (short)0, c, false, false);
}

DEV float sigf(float x)  { return 1.0f / (1.0f + __expf(-x)); }

// ---------------------------------------------------------------------------
// Block reductions / scan (blockDim.x == 256, 8 waves of 32)
// ---------------------------------------------------------------------------
DEV float blk_sum(float v, float* sh) {
    for (int off = 16; off > 0; off >>= 1) v += __shfl_xor(v, off, 32);
    int lane = threadIdx.x & 31, wid = threadIdx.x >> 5;
    if (lane == 0) sh[wid] = v;
    __syncthreads();
    if (wid == 0) {
        float r = (lane < 8) ? sh[lane] : 0.0f;
        for (int off = 4; off > 0; off >>= 1) r += __shfl_xor(r, off, 32);
        if (lane == 0) sh[0] = r;
    }
    __syncthreads();
    float out = sh[0];
    __syncthreads();
    return out;
}

DEV float blk_max(float v, float* sh) {
    for (int off = 16; off > 0; off >>= 1) v = fmaxf(v, __shfl_xor(v, off, 32));
    int lane = threadIdx.x & 31, wid = threadIdx.x >> 5;
    if (lane == 0) sh[wid] = v;
    __syncthreads();
    if (wid == 0) {
        float r = (lane < 8) ? sh[lane] : -INFINITY;
        for (int off = 4; off > 0; off >>= 1) r = fmaxf(r, __shfl_xor(r, off, 32));
        if (lane == 0) sh[0] = r;
    }
    __syncthreads();
    float out = sh[0];
    __syncthreads();
    return out;
}

// exclusive prefix over per-thread values (256 threads)
DEV float blk_scan_excl(float v, float* sh) {
    int lane = threadIdx.x & 31, wid = threadIdx.x >> 5;
    float x = v;
    for (int off = 1; off < 32; off <<= 1) {
        float y = __shfl_up(x, off, 32);
        if (lane >= off) x += y;
    }
    if (lane == 31) sh[wid] = x;
    __syncthreads();
    if (threadIdx.x == 0) {
        float run = 0.0f;
        for (int w = 0; w < 8; ++w) { float t = sh[w]; sh[w] = run; run += t; }
    }
    __syncthreads();
    float excl = (x - v) + sh[wid];
    __syncthreads();
    return excl;
}

// ---------------------------------------------------------------------------
// One-time prep kernels
// ---------------------------------------------------------------------------
__global__ void dot6_kernel(const float* a0, const float* b0, const float* a1, const float* b1,
                            const float* a2, const float* b2, const float* a3, const float* b3,
                            const float* a4, const float* b4, const float* a5, const float* b5,
                            float* s) {
    __shared__ float sh[8];
    const float* as[6] = {a0, a1, a2, a3, a4, a5};
    const float* bs[6] = {b0, b1, b2, b3, b4, b5};
    for (int j = 0; j < 6; ++j) {
        float v = 0.0f;
        for (int i = threadIdx.x; i < 1024; i += 256) v += as[j][i] * bs[j][i];
        v = blk_sum(v, sh);
        if (threadIdx.x == 0) s[j] = v;
        __syncthreads();
    }
}

// dst[c*R + r] = bf16(src[r*C + c])   (src is R x C row-major -> dst is C x R)
__global__ void transpose_bf16_kernel(const float* __restrict__ src, __bf16* __restrict__ dst,
                                      int R, int C) {
    long total = (long)R * C;
    for (long i = blockIdx.x * (long)blockDim.x + threadIdx.x; i < total;
         i += (long)gridDim.x * blockDim.x) {
        int r = (int)(i / C);
        int c = (int)(i % C);
        dst[(long)c * R + r] = (__bf16)src[i];
    }
}

__global__ void convert_bf16_kernel(const float* __restrict__ src, __bf16* __restrict__ dst,
                                    long n) {
    for (long i = blockIdx.x * (long)blockDim.x + threadIdx.x; i < n;
         i += (long)gridDim.x * blockDim.x)
        dst[i] = (__bf16)src[i];
}

__global__ void init_state_kernel(__bf16* hbf, __bf16* cbf, float* cf) {
    int i = blockIdx.x * blockDim.x + threadIdx.x;
    if (i < B * U) {
        hbf[i] = (__bf16)0.0f;
        cbf[i] = (__bf16)0.0f;
        cf[i]  = 0.0f;
    }
}

// ---------------------------------------------------------------------------
// Step kernel 1: z = bias + x_t @ K + h @ R + c @ C        (64 x 5120, K=2304)
// grid (40, 4), block 128 (4 waves); wave -> 16x32 output (2 N-tiles share A).
// Inner loop is explicitly software-pipelined (double-buffered fragments) so
// WMMAs overlap the next iteration's global_load_b128 clauses.
// ---------------------------------------------------------------------------
__global__ void step_z_kernel(const __bf16* __restrict__ xbf,    // [B][T][D]
                              const __bf16* __restrict__ kern_t, // [5U][D]
                              const __bf16* __restrict__ rk_t,   // [5U][U]
                              const __bf16* __restrict__ ck_t,   // [5U][U]
                              const __bf16* __restrict__ hbf,    // [B][U]
                              const __bf16* __restrict__ cbf,    // [B][U]
                              const float*  __restrict__ bias,   // [5U]
                              float* __restrict__ z,             // [B][5U]
                              int t) {
    const int wave = threadIdx.x >> 5;
    const int lane = threadIdx.x & 31;
    const int half = lane >> 4;
    const int ln   = lane & 15;
    const int nBase0 = (blockIdx.x * 4 + wave) * 32;  // first N tile
    const int nBase1 = nBase0 + 16;                   // second N tile
    const int mBase  = blockIdx.y * 16;               // 0..48

    // A rows (shared by both N tiles)
    const __bf16* xrow  = xbf + (long)(mBase + ln) * (T * D) + (long)t * D;
    const __bf16* hrow  = hbf + (long)(mBase + ln) * U;
    const __bf16* crow  = cbf + (long)(mBase + ln) * U;
    // B rows for the two N tiles
    const __bf16* krow0 = kern_t + (long)(nBase0 + ln) * D;
    const __bf16* krow1 = kern_t + (long)(nBase1 + ln) * D;
    const __bf16* rrow0 = rk_t + (long)(nBase0 + ln) * U;
    const __bf16* rrow1 = rk_t + (long)(nBase1 + ln) * U;
    const __bf16* wrow0 = ck_t + (long)(nBase0 + ln) * U;
    const __bf16* wrow1 = ck_t + (long)(nBase1 + ln) * U;

    v8f acc0 = {};
    v8f acc1 = {};

    // ---- x_t @ kernel (K = 256), 2-deep pipeline ----
    {
        v16bf ax = ldfrag(xrow, 0, half);
        v16bf b0 = ldfrag(krow0, 0, half);
        v16bf b1 = ldfrag(krow1, 0, half);
        for (int k0 = 0; k0 < D - 32; k0 += 32) {
            v16bf ax_n = ldfrag(xrow, k0 + 32, half);
            v16bf b0_n = ldfrag(krow0, k0 + 32, half);
            v16bf b1_n = ldfrag(krow1, k0 + 32, half);
            acc0 = wmma_bf16(ax, b0, acc0);
            acc1 = wmma_bf16(ax, b1, acc1);
            ax = ax_n; b0 = b0_n; b1 = b1_n;
        }
        acc0 = wmma_bf16(ax, b0, acc0);
        acc1 = wmma_bf16(ax, b1, acc1);
    }

    // ---- h @ R + c @ C (K = 1024 each), 2-deep pipeline ----
    {
        v16bf ah  = ldfrag(hrow, 0, half);
        v16bf br0 = ldfrag(rrow0, 0, half);
        v16bf br1 = ldfrag(rrow1, 0, half);
        v16bf ac  = ldfrag(crow, 0, half);
        v16bf bw0 = ldfrag(wrow0, 0, half);
        v16bf bw1 = ldfrag(wrow1, 0, half);
        for (int k0 = 0; k0 < U - 32; k0 += 32) {
            const int kn = k0 + 32;
            // speculative prefetch two iterations ahead (dropped if OOB)
            __builtin_prefetch((const void*)(rrow0 + k0 + 64), 0, 3);
            __builtin_prefetch((const void*)(wrow0 + k0 + 64), 0, 3);
            v16bf ah_n  = ldfrag(hrow, kn, half);
            v16bf br0_n = ldfrag(rrow0, kn, half);
            v16bf br1_n = ldfrag(rrow1, kn, half);
            v16bf ac_n  = ldfrag(crow, kn, half);
            v16bf bw0_n = ldfrag(wrow0, kn, half);
            v16bf bw1_n = ldfrag(wrow1, kn, half);
            acc0 = wmma_bf16(ah, br0, acc0);
            acc1 = wmma_bf16(ah, br1, acc1);
            acc0 = wmma_bf16(ac, bw0, acc0);
            acc1 = wmma_bf16(ac, bw1, acc1);
            ah = ah_n; br0 = br0_n; br1 = br1_n;
            ac = ac_n; bw0 = bw0_n; bw1 = bw1_n;
        }
        acc0 = wmma_bf16(ah, br0, acc0);
        acc1 = wmma_bf16(ah, br1, acc1);
        acc0 = wmma_bf16(ac, bw0, acc0);
        acc1 = wmma_bf16(ac, bw1, acc1);
    }

    const float bn0 = bias[nBase0 + ln];
    const float bn1 = bias[nBase1 + ln];
#pragma unroll
    for (int v = 0; v < 8; ++v) {
        int m = mBase + v + 8 * half;                 // C/D layout: M = v + 8*half
        z[(long)m * U5 + nBase0 + ln] = acc0[v] + bn0;
        z[(long)m * U5 + nBase1 + ln] = acc1[v] + bn1;
    }
}

// ---------------------------------------------------------------------------
// Step kernel 2: gates. One block per batch row; 256 threads * 4 elems = 1024.
// ---------------------------------------------------------------------------
__global__ void step_gate_kernel(const float* __restrict__ z,   // [B][5U]
                                 const float* __restrict__ s,   // [6]
                                 __bf16* __restrict__ Obf) {    // [B][3U]
    __shared__ float sh[8];
    const int b = blockIdx.x;
    const int base = threadIdx.x * 4;
    const float* zrow = z + (long)b * U5;

    float uz[4], dz[4], rv[4];
#pragma unroll
    for (int i = 0; i < 4; ++i) {
        uz[i] = zrow[base + i];
        dz[i] = zrow[U + base + i];
        rv[i] = zrow[2 * U + base + i];
    }

    // cumsoftmax left-to-right over uz
    float mx = fmaxf(fmaxf(uz[0], uz[1]), fmaxf(uz[2], uz[3]));
    mx = blk_max(mx, sh);
    float eu[4], lsu = 0.0f;
#pragma unroll
    for (int i = 0; i < 4; ++i) { eu[i] = __expf(uz[i] - mx); lsu += eu[i]; }
    float totu = blk_sum(lsu, sh);
    float exu  = blk_scan_excl(lsu, sh);
    float invu = 1.0f / totu;
    float up[4], run = exu;
#pragma unroll
    for (int i = 0; i < 4; ++i) { run += eu[i]; up[i] = run * invu; }

    // cumsoftmax right-to-left over dz: down[i] = 1 - (cum_incl(i) - p_i)
    float mx2 = fmaxf(fmaxf(dz[0], dz[1]), fmaxf(dz[2], dz[3]));
    mx2 = blk_max(mx2, sh);
    float ed[4], lsd = 0.0f;
#pragma unroll
    for (int i = 0; i < 4; ++i) { ed[i] = __expf(dz[i] - mx2); lsd += ed[i]; }
    float totd = blk_sum(lsd, sh);
    float exd  = blk_scan_excl(lsd, sh);
    float invd = 1.0f / totd;
    float dn[4], run2 = exd;
#pragma unroll
    for (int i = 0; i < 4; ++i) { run2 += ed[i]; dn[i] = 1.0f - (run2 - ed[i]) * invd; }

    // row reductions
    float ldu = 0.0f, lru = 0.0f, ldr = 0.0f;
#pragma unroll
    for (int i = 0; i < 4; ++i) {
        ldu += dn[i] * up[i];
        lru += rv[i] * up[i];
        ldr += dn[i] * rv[i];
    }
    float du = blk_sum(ldu, sh);
    float ru = blk_sum(lru, sh);
    float dr = blk_sum(ldr, sh);

    const float s_ud = s[0], s_ur = s[1], s_ru = s[2], s_rd = s[3], s_du = s[4], s_dr = s[5];
    __bf16* Orow = Obf + (long)b * U3;
#pragma unroll
    for (int i = 0; i < 4; ++i) {
        int idx = base + i;
        float O1 = sigf(s_ud * up[i] * du) + sigf(s_ur * up[i] * ru);
        float O2 = sigf(s_ru * rv[i] * ru) + sigf(s_rd * rv[i] * dr);
        float O3 = sigf(s_du * dn[i] * du) + sigf(s_dr * dn[i] * dr);
        Orow[idx]         = (__bf16)O1;
        Orow[U + idx]     = (__bf16)O2;
        Orow[2 * U + idx] = (__bf16)O3;
    }
}

// ---------------------------------------------------------------------------
// Step kernel 3: f = sigmoid(O @ agg), fused c/h update.   (64 x 1024, K=3072)
// grid (16, 4), block 128 (4 waves); wave -> one 16x16 tile, pipelined.
// ---------------------------------------------------------------------------
__global__ void step_agg_kernel(const __bf16* __restrict__ Obf,   // [B][3U]
                                const __bf16* __restrict__ agg_t, // [U][3U]
                                const float*  __restrict__ z,     // [B][5U]
                                float* __restrict__ cstate,       // [B][U]
                                __bf16* __restrict__ hbf,
                                __bf16* __restrict__ cbf,
                                float* __restrict__ out,          // [B][T][U]
                                int t) {
    const int wave = threadIdx.x >> 5;
    const int lane = threadIdx.x & 31;
    const int half = lane >> 4;
    const int ln   = lane & 15;
    const int nBase = (blockIdx.x * 4 + wave) * 16;
    const int mBase = blockIdx.y * 16;

    const __bf16* arow = Obf   + (long)(mBase + ln) * U3;
    const __bf16* brow = agg_t + (long)(nBase + ln) * U3;

    v8f acc = {};
    {
        v16bf a = ldfrag(arow, 0, half);
        v16bf b = ldfrag(brow, 0, half);
        for (int k0 = 0; k0 < U3 - 32; k0 += 32) {
            __builtin_prefetch((const void*)(brow + k0 + 64), 0, 3);
            v16bf a_n = ldfrag(arow, k0 + 32, half);
            v16bf b_n = ldfrag(brow, k0 + 32, half);
            acc = wmma_bf16(a, b, acc);
            a = a_n; b = b_n;
        }
        acc = wmma_bf16(a, b, acc);
    }

#pragma unroll
    for (int v = 0; v < 8; ++v) {
        int m = mBase + v + 8 * half;
        int n = nBase + ln;
        float f    = sigf(acc[v]);
        float cold = cstate[(long)m * U + n];
        float cin  = tanhf(z[(long)m * U5 + 4 * U + n]);
        float cn   = f * cold + (1.0f - f) * cin;
        float og   = z[(long)m * U5 + 3 * U + n];   // o_gate is raw (no sigmoid)
        float hn   = og * tanhf(cn);
        cstate[(long)m * U + n] = cn;
        cbf[(long)m * U + n] = (__bf16)cn;
        hbf[(long)m * U + n] = (__bf16)hn;
        out[(long)m * T * U + (long)t * U + n] = hn;
    }
}

// ---------------------------------------------------------------------------
// Host launch
// ---------------------------------------------------------------------------
extern "C" void kernel_launch(void* const* d_in, const int* in_sizes, int n_in,
                              void* d_out, int out_size, void* d_ws, size_t ws_size,
                              hipStream_t stream) {
    const float* x    = (const float*)d_in[0];
    const float* kern = (const float*)d_in[1];
    const float* rk   = (const float*)d_in[2];
    const float* ck   = (const float*)d_in[3];
    const float* bias = (const float*)d_in[4];
    const float* agg  = (const float*)d_in[5];
    // attention vectors: d_in[6..17]

    char* w = (char*)d_ws;
    size_t off = 0;
    auto alloc = [&](size_t bytes) -> void* {
        void* p = w + off;
        off = (off + bytes + 255) & ~(size_t)255;
        return p;
    };
    float*  s6     = (float*) alloc(6 * sizeof(float));
    __bf16* kern_t = (__bf16*)alloc((size_t)U5 * D  * sizeof(__bf16));
    __bf16* rk_t   = (__bf16*)alloc((size_t)U5 * U  * sizeof(__bf16));
    __bf16* ck_t   = (__bf16*)alloc((size_t)U5 * U  * sizeof(__bf16));
    __bf16* agg_t  = (__bf16*)alloc((size_t)U  * U3 * sizeof(__bf16));
    __bf16* xbf    = (__bf16*)alloc((size_t)B * T * D * sizeof(__bf16));
    float*  zbuf   = (float*) alloc((size_t)B * U5 * sizeof(float));
    __bf16* Obf    = (__bf16*)alloc((size_t)B * U3 * sizeof(__bf16));
    __bf16* hbf    = (__bf16*)alloc((size_t)B * U * sizeof(__bf16));
    __bf16* cbf    = (__bf16*)alloc((size_t)B * U * sizeof(__bf16));
    float*  cf32   = (float*) alloc((size_t)B * U * sizeof(float));
    (void)ws_size;

    float* out = (float*)d_out;

    // attention scalars
    dot6_kernel<<<1, 256, 0, stream>>>(
        (const float*)d_in[6],  (const float*)d_in[7],
        (const float*)d_in[8],  (const float*)d_in[9],
        (const float*)d_in[10], (const float*)d_in[11],
        (const float*)d_in[12], (const float*)d_in[13],
        (const float*)d_in[14], (const float*)d_in[15],
        (const float*)d_in[16], (const float*)d_in[17], s6);

    // weight transpose + bf16 conversion (N-major for WMMA B fragments)
    transpose_bf16_kernel<<<4096, 256, 0, stream>>>(kern, kern_t, D,  U5);
    transpose_bf16_kernel<<<8192, 256, 0, stream>>>(rk,   rk_t,   U,  U5);
    transpose_bf16_kernel<<<8192, 256, 0, stream>>>(ck,   ck_t,   U,  U5);
    transpose_bf16_kernel<<<8192, 256, 0, stream>>>(agg,  agg_t,  U3, U);
    convert_bf16_kernel<<<4096, 256, 0, stream>>>(x, xbf, (long)B * T * D);
    init_state_kernel<<<(B * U + 255) / 256, 256, 0, stream>>>(hbf, cbf, cf32);

    // sequential scan over time
    for (int t = 0; t < T; ++t) {
        step_z_kernel<<<dim3(40, 4), 128, 0, stream>>>(
            xbf, kern_t, rk_t, ck_t, hbf, cbf, bias, zbuf, t);
        step_gate_kernel<<<B, 256, 0, stream>>>(zbuf, s6, Obf);
        step_agg_kernel<<<dim3(16, 4), 128, 0, stream>>>(
            Obf, agg_t, zbuf, cf32, hbf, cbf, out, t);
    }
}
